// AutoregressiveTransformerModel_31748398252899
// MI455X (gfx1250) — matrix operand order repaired and checked by reading, not compiled, not run on hardware
//
#include <hip/hip_runtime.h>
#include <hip/hip_bf16.h>
#include <stdint.h>

// ---------------------------------------------------------------------------
// AutoregressiveTransformerModel forward for MI455X (gfx1250, wave32, WMMA)
// B=4, S=1024, D=512, H=8, HD=64, L=12, DFF=2048, SCHEMA=21, VOCAB=19
// ---------------------------------------------------------------------------

#define BB   4
#define SS   1024
#define DD   512
#define HH   8
#define HDIM 64
#define LL   12
#define DFFN 2048
#define TOK  (BB * SS)        // 4096 tokens
#define VOC  19
#define SCH  21

typedef __attribute__((ext_vector_type(16))) _Float16 v16h;
typedef __attribute__((ext_vector_type(8)))  _Float16 v8h;
typedef __attribute__((ext_vector_type(8)))  float    v8f;

union FragU { v16h v; v8h h[2]; };

__device__ inline v8f zero8() {
  v8f z = {0.f,0.f,0.f,0.f,0.f,0.f,0.f,0.f};
  return z;
}

// A fragment (16x32 f16, MxK). rowptr -> start of this row's 32-wide K chunk.
// lane hi=0: K = {0..15};  hi=1: K = {16..31}
__device__ inline v16h load_frag_a(const _Float16* rowptr, int hi) {
  FragU f;
  f.h[0] = *(const v8h*)(rowptr + 16 * hi);
  f.h[1] = *(const v8h*)(rowptr + 16 * hi + 8);
  return f.v;
}

// B fragment (32x16 f16, KxN), column-contiguous source (N-major "Bt").
// colptr -> start of this column's 32-wide K chunk.
// lane hi=0: K = {0..15};  hi=1: K = {16..31}
__device__ inline v16h load_frag_b(const _Float16* colptr, int hi) {
  FragU f;
  const _Float16* p = colptr + 16 * hi;
  f.h[0] = *(const v8h*)(p);
  f.h[1] = *(const v8h*)(p + 8);
  return f.v;
}

__device__ inline v8f wmma16(v16h a, v16h b, v8f c) {
  return __builtin_amdgcn_wmma_f32_16x16x32_f16(
      /*neg_a=*/false, a, /*neg_b=*/false, b,
      /*c_mod=*/(short)0, c, /*reuse_a=*/false, /*reuse_b=*/false);
}

__device__ inline float gelu_f(float x) {
  float x3 = x * x * x;
  return 0.5f * x * (1.f + tanhf(0.7978845608028654f * (x + 0.044715f * x3)));
}

// ---------------------------------------------------------------------------
// Weight convert + transpose: out[c*R + r] = (f16) in[r*C + c], per layer (grid.y)
// ---------------------------------------------------------------------------
__global__ void tc_kernel(const float* __restrict__ in, _Float16* __restrict__ out,
                          int R, int C) {
  size_t base = (size_t)blockIdx.y * R * C;
  int idx = blockIdx.x * 256 + threadIdx.x;
  if (idx >= R * C) return;
  int r = idx / C, c = idx - r * C;
  out[base + (size_t)c * R + r] = (_Float16)in[base + idx];
}

// ---------------------------------------------------------------------------
// Embedding: h = xs @ read_w + read_b + pos
// ---------------------------------------------------------------------------
__global__ __launch_bounds__(256) void embed_kernel(
    const float* __restrict__ xs, const float* __restrict__ rw,
    const float* __restrict__ rb, const float* __restrict__ ps,
    const float* __restrict__ pp, float* __restrict__ h) {
  __shared__ float xr[64];
  int t = blockIdx.x;            // token 0..4095
  int s = t & (SS - 1);
  if (threadIdx.x < 64) xr[threadIdx.x] = xs[(size_t)t * 64 + threadIdx.x];
  __syncthreads();
  for (int d = threadIdx.x; d < DD; d += 256) {
    float acc = rb[d] + (s < SCH ? ps[s * DD + d] : pp[(s - SCH) * DD + d]);
    #pragma unroll 8
    for (int k = 0; k < 64; k++) acc += xr[k] * rw[k * DD + d];
    h[(size_t)t * DD + d] = acc;
  }
}

// ---------------------------------------------------------------------------
// LayerNorm: one block per token row of 512
// ---------------------------------------------------------------------------
__global__ __launch_bounds__(256) void ln_kernel(
    const float* __restrict__ h, const float* __restrict__ g,
    const float* __restrict__ b, _Float16* __restrict__ out16,
    float* __restrict__ out32, int wideF32) {
  __shared__ float red[256];
  int t = blockIdx.x;
  const float* row = h + (size_t)t * DD;
  float x0 = row[threadIdx.x], x1 = row[threadIdx.x + 256];
  red[threadIdx.x] = x0 + x1;
  __syncthreads();
  for (int off = 128; off > 0; off >>= 1) {
    if (threadIdx.x < off) red[threadIdx.x] += red[threadIdx.x + off];
    __syncthreads();
  }
  float mu = red[0] * (1.f / DD);
  __syncthreads();
  float d0 = x0 - mu, d1 = x1 - mu;
  red[threadIdx.x] = d0 * d0 + d1 * d1;
  __syncthreads();
  for (int off = 128; off > 0; off >>= 1) {
    if (threadIdx.x < off) red[threadIdx.x] += red[threadIdx.x + off];
    __syncthreads();
  }
  float rstd = rsqrtf(red[0] * (1.f / DD) + 1e-5f);
  float y0 = d0 * rstd * g[threadIdx.x] + b[threadIdx.x];
  float y1 = d1 * rstd * g[threadIdx.x + 256] + b[threadIdx.x + 256];
  if (wideF32) {
    out32[(size_t)t * DD + threadIdx.x] = y0;
    out32[(size_t)t * DD + threadIdx.x + 256] = y1;
  } else {
    out16[(size_t)t * DD + threadIdx.x] = (_Float16)y0;
    out16[(size_t)t * DD + threadIdx.x + 256] = (_Float16)y1;
  }
}

// ---------------------------------------------------------------------------
// WMMA GEMM with CDNA5 async-LDS B staging.
// C[M,N] = A[M,K](f16) * Bt[N,K](f16, N-major) + bias
// 128 threads = 4 waves; wave tile 32(M) x 64(N); block tile 128(M) x 64(N).
// B 64x32 tile staged to LDS via global_load_async_to_lds_b128 (ASYNCcnt),
// double-buffered; fragments read back as conflict-free ds_load_b128
// (per-column LDS row padded to 80B -> 16 disjoint bank groups).
// mode 0: outH = acc+bias (f16) | mode 1: outH = gelu(acc+bias) (f16)
// mode 2: outF += acc+bias (f32 residual accumulate)
// ---------------------------------------------------------------------------
#define BPAD 40   // halves per LDS column row (80 bytes)

__global__ __launch_bounds__(128) void gemm_wmma(
    const _Float16* __restrict__ A, const _Float16* __restrict__ Bt,
    const float* __restrict__ bias, _Float16* __restrict__ outH,
    float* __restrict__ outF, int M, int N, int K, int mode) {
  __shared__ _Float16 Bs[2][64 * BPAD];
  int tid  = threadIdx.x;
  int wave = tid >> 5, lane = tid & 31;
  int l16 = lane & 15, hi = lane >> 4;
  int mt = blockIdx.y * 128 + wave * 32;
  int nb = blockIdx.x * 64;

  const _Float16* arow0 = A + (size_t)(mt + l16) * K;
  const _Float16* arow1 = A + (size_t)(mt + 16 + l16) * K;

  // Staging: 64 cols x 64B = 256 16-byte chunks; thread covers chunks tid, tid+128.
  int c0 = tid >> 2, q0 = tid & 3;                     // chunk tid
  const _Float16* g0 = Bt + (size_t)(nb + c0) * K + q0 * 8;
  const _Float16* g1 = Bt + (size_t)(nb + c0 + 32) * K + q0 * 8;
  uint32_t lds0 = (uint32_t)(uintptr_t)(&Bs[0][0]);
  uint32_t la0 = lds0 + (uint32_t)(c0 * (BPAD * 2) + q0 * 16);
  uint32_t lb0 = la0 + 32u * BPAD * 2;
  uint32_t bufstride = (uint32_t)(64 * BPAD * 2);

  v8f acc[2][4];
  #pragma unroll
  for (int g = 0; g < 2; g++)
    #pragma unroll
    for (int s = 0; s < 4; s++) acc[g][s] = zero8();

  auto stage = [&](int k, int buf) {
    uint64_t ga = (uint64_t)(uintptr_t)(g0 + k);
    uint64_t gb = (uint64_t)(uintptr_t)(g1 + k);
    uint32_t la = la0 + (uint32_t)buf * bufstride;
    uint32_t lb = lb0 + (uint32_t)buf * bufstride;
    asm volatile(
        "global_load_async_to_lds_b128 %0, %1, off\n\t"
        "global_load_async_to_lds_b128 %2, %3, off"
        :: "v"(la), "v"(ga), "v"(lb), "v"(gb) : "memory");
  };

  int nsteps = K >> 5;
  stage(0, 0);
  for (int i = 0; i < nsteps; i++) {
    int k = i << 5;
    int buf = i & 1;
    asm volatile("s_wait_asynccnt 0x0" ::: "memory");
    __syncthreads();                       // staged tile visible to all waves
    if (i + 1 < nsteps) stage(k + 32, buf ^ 1);
    if (k + 64 < K) __builtin_prefetch(arow0 + k + 64, 0, 1);
    v16h a0 = load_frag_a(arow0 + k, hi);
    v16h a1 = load_frag_a(arow1 + k, hi);
    const _Float16* Bsb = &Bs[buf][0];
    #pragma unroll
    for (int s = 0; s < 4; s++) {
      FragU f;
      const _Float16* base = Bsb + (s * 16 + l16) * BPAD + 16 * hi;
      f.h[0] = *(const v8h*)(base);
      f.h[1] = *(const v8h*)(base + 8);
      acc[0][s] = wmma16(a0, f.v, acc[0][s]);
      acc[1][s] = wmma16(a1, f.v, acc[1][s]);
    }
    // No trailing barrier needed: next write to this buffer happens after
    // the *next* iteration's top barrier (double-buffered).
  }

  #pragma unroll
  for (int g = 0; g < 2; g++) {
    #pragma unroll
    for (int s = 0; s < 4; s++) {
      int col = nb + s * 16 + l16;
      float bv = bias[col];
      #pragma unroll
      for (int r = 0; r < 8; r++) {
        int row = mt + g * 16 + r + 8 * hi;
        float v = acc[g][s][r] + bv;
        size_t idx = (size_t)row * N + col;
        if (mode == 0)      outH[idx] = (_Float16)v;
        else if (mode == 1) outH[idx] = (_Float16)gelu_f(v);
        else                outF[idx] = outF[idx] + v;
      }
    }
  }
}

// ---------------------------------------------------------------------------
// V transpose per head via LDS tile (both sides coalesced):
// vT[(b*H+h)*64 + d][s] = qkv[b,s, 2D + h*64 + d]
// grid: (S/64, B*H); block 256
// ---------------------------------------------------------------------------
__global__ __launch_bounds__(256) void vt_kernel(const _Float16* __restrict__ qkv,
                                                 _Float16* __restrict__ vT) {
  __shared__ _Float16 tile[64][72];   // pad 72 halves to dodge bank conflicts
  int bh = blockIdx.y;                // 0..31
  int b = bh >> 3, hh = bh & 7;
  int s0 = blockIdx.x * 64;
  int td = threadIdx.x & 63;          // d on load, s on store
  int tr = threadIdx.x >> 6;          // 0..3
  #pragma unroll
  for (int i = 0; i < 64; i += 4) {
    int s = s0 + tr + i;
    tile[tr + i][td] =
        qkv[(size_t)(b * SS + s) * (3 * DD) + 2 * DD + hh * HDIM + td];
  }
  __syncthreads();
  #pragma unroll
  for (int i = 0; i < 64; i += 4) {
    int d = tr + i;
    vT[((size_t)bh * HDIM + d) * SS + s0 + td] = tile[td][d];
  }
}

// ---------------------------------------------------------------------------
// Flash attention, one wave per (b, head, 16-query tile).
// ---------------------------------------------------------------------------
__device__ inline bool attn_allowed(int q, int k, int use_first) {
  if (k > q) return false;            // causal
  if (!use_first) return true;        // mask_rest == causal
  if (q < 20)  return (k >> 2) == (q >> 2);  // schema blocks of (C_+1)=4
  if (q == 20) return true;           // summary row attends all schema
  return k >= SCH;                    // path rows attend path only
}

__global__ __launch_bounds__(128) void attn_kernel(
    const _Float16* __restrict__ qkv, const _Float16* __restrict__ vT,
    _Float16* __restrict__ ctx, int use_first) {
  __shared__ _Float16 Plds[4][16 * 32];
  int wave = threadIdx.x >> 5;
  int lane = threadIdx.x & 31;
  int unit = blockIdx.x * 4 + wave;               // 0..2047
  int qt = unit & 63;
  int hh = (unit >> 6) & 7;
  int b  = unit >> 9;
  int qbase = qt * 16;
  int l16 = lane & 15, hi = lane >> 4;

  const _Float16* qrow =
      qkv + ((size_t)(b * SS) + qbase + l16) * (3 * DD) + hh * HDIM;
  v16h qa0 = load_frag_a(qrow, hi);       // hd 0..31
  v16h qa1 = load_frag_a(qrow + 32, hi);  // hd 32..63

  v8f acc[4];
  #pragma unroll
  for (int s = 0; s < 4; s++) acc[s] = zero8();
  float mrow[8], lrow[8];
  #pragma unroll
  for (int r = 0; r < 8; r++) { mrow[r] = -1e28f; lrow[r] = 0.f; }

  _Float16* Pw = Plds[wave];
  const _Float16* vhead = vT + ((size_t)(b * HH + hh) * HDIM + l16) * SS;
  int kend = qbase + 16;                  // causal upper bound (exclusive)

  for (int kc = 0; kc < kend; kc += 32) {
    // -------- scores: two 16x16 tiles (keys kc..kc+15, kc+16..kc+31) ------
    v8f s0 = zero8(), s1 = zero8();
    {
      const _Float16* kcol0 =
          qkv + ((size_t)(b * SS) + kc + l16) * (3 * DD) + DD + hh * HDIM;
      s0 = wmma16(qa0, load_frag_b(kcol0, hi), s0);
      s0 = wmma16(qa1, load_frag_b(kcol0 + 32, hi), s0);
      const _Float16* kcol1 = kcol0 + (size_t)16 * (3 * DD);
      s1 = wmma16(qa0, load_frag_b(kcol1, hi), s1);
      s1 = wmma16(qa1, load_frag_b(kcol1 + 32, hi), s1);
    }
    // -------- mask + online softmax (rows live per-VGPR, per 16-lane half) -
    float p0[8], p1[8];
    #pragma unroll
    for (int r = 0; r < 8; r++) {
      int q  = qbase + r + 8 * hi;
      int k0 = kc + l16, k1 = k0 + 16;
      float v0 = attn_allowed(q, k0, use_first) ? s0[r] * 0.125f : -1e30f;
      float v1 = attn_allowed(q, k1, use_first) ? s1[r] * 0.125f : -1e30f;
      float mx = fmaxf(v0, v1);
      for (int off = 1; off < 16; off <<= 1)
        mx = fmaxf(mx, __shfl_xor(mx, off, 16));
      float nm = fmaxf(mrow[r], mx);
      float e0 = __expf(v0 - nm), e1 = __expf(v1 - nm);
      float rs = e0 + e1;
      for (int off = 1; off < 16; off <<= 1) rs += __shfl_xor(rs, off, 16);
      float alpha = __expf(mrow[r] - nm);
      lrow[r] = lrow[r] * alpha + rs;
      mrow[r] = nm;
      p0[r] = e0; p1[r] = e1;
      acc[0][r] *= alpha; acc[1][r] *= alpha;
      acc[2][r] *= alpha; acc[3][r] *= alpha;
    }
    // -------- stage P through LDS: C-layout -> A-layout re-swizzle --------
    #pragma unroll
    for (int r = 0; r < 8; r++) {
      int row = r + 8 * hi;
      Pw[row * 32 + l16]      = (_Float16)p0[r];
      Pw[row * 32 + 16 + l16] = (_Float16)p1[r];
    }
    v16h pa;
    {
      FragU f;
      const _Float16* base = Pw + l16 * 32;
      f.h[0] = *(const v8h*)(base + 16 * hi);
      f.h[1] = *(const v8h*)(base + 16 * hi + 8);
      pa = f.v;
    }
    // -------- ctx += P(16x32) x V(32x64) ----------------------------------
    #pragma unroll
    for (int s = 0; s < 4; s++) {
      v16h vb = load_frag_b(vhead + (size_t)s * 16 * SS + kc, hi);
      acc[s] = wmma16(pa, vb, acc[s]);
    }
  }

  // -------- finalize: divide by row sums, store f16 ------------------------
  #pragma unroll
  for (int s = 0; s < 4; s++) {
    #pragma unroll
    for (int r = 0; r < 8; r++) {
      int row = qbase + r + 8 * hi;
      float val = acc[s][r] / lrow[r];
      ctx[(size_t)(b * SS + row) * DD + hh * HDIM + s * 16 + l16] =
          (_Float16)val;
    }
  }
}

// ---------------------------------------------------------------------------
// Output head: logits = xf @ out_w + out_b (512 x 19, scalar)
// ---------------------------------------------------------------------------
__global__ void head_kernel(const float* __restrict__ xf,
                            const float* __restrict__ ow,
                            const float* __restrict__ ob,
                            float* __restrict__ out) {
  int t = blockIdx.x;
  int v = threadIdx.x;
  if (v >= VOC) return;
  float acc = ob[v];
  const float* row = xf + (size_t)t * DD;
  for (int k = 0; k < DD; k++) acc += row[k] * ow[k * VOC + v];
  out[(size_t)t * VOC + v] = acc;
}

// ---------------------------------------------------------------------------
extern "C" void kernel_launch(void* const* d_in, const int* in_sizes, int n_in,
                              void* d_out, int out_size, void* d_ws,
                              size_t ws_size, hipStream_t stream) {
  (void)in_sizes; (void)n_in; (void)out_size; (void)ws_size;
  const float* xs       = (const float*)d_in[0];
  const float* read_w   = (const float*)d_in[1];
  const float* read_b   = (const float*)d_in[2];
  const float* pos_sch  = (const float*)d_in[3];
  const float* pos_path = (const float*)d_in[4];
  const float* ln1_g    = (const float*)d_in[5];
  const float* ln1_b    = (const float*)d_in[6];
  const float* attn_w   = (const float*)d_in[7];
  const float* attn_b   = (const float*)d_in[8];
  const float* attnp_w  = (const float*)d_in[9];
  const float* attnp_b  = (const float*)d_in[10];
  const float* ln2_g    = (const float*)d_in[11];
  const float* ln2_b    = (const float*)d_in[12];
  const float* fc_w     = (const float*)d_in[13];
  const float* fc_b     = (const float*)d_in[14];
  const float* proj_w   = (const float*)d_in[15];
  const float* proj_b   = (const float*)d_in[16];
  const float* lnf_g    = (const float*)d_in[17];
  const float* lnf_b    = (const float*)d_in[18];
  const float* out_w    = (const float*)d_in[19];
  const float* out_b    = (const float*)d_in[20];

  char* w = (char*)d_ws;
  auto alloc = [&](size_t bytes) {
    void* p = (void*)w;
    w += (bytes + 255) & ~(size_t)255;
    return p;
  };
  _Float16* wqT  = (_Float16*)alloc((size_t)LL * 3 * DD * DD * 2);   // [l][1536][512]
  _Float16* wpT  = (_Float16*)alloc((size_t)LL * DD * DD * 2);       // [l][512][512]
  _Float16* wfT  = (_Float16*)alloc((size_t)LL * DFFN * DD * 2);     // [l][2048][512]
  _Float16* wprT = (_Float16*)alloc((size_t)LL * DD * DFFN * 2);     // [l][512][2048]
  float*    hbuf = (float*)   alloc((size_t)TOK * DD * 4);
  _Float16* x16  = (_Float16*)alloc((size_t)TOK * DD * 2);
  _Float16* c16  = (_Float16*)alloc((size_t)TOK * DD * 2);
  _Float16* qkv16= (_Float16*)alloc((size_t)TOK * 3 * DD * 2);
  _Float16* vT16 = (_Float16*)alloc((size_t)BB * HH * HDIM * SS * 2);
  _Float16* ff16 = (_Float16*)alloc((size_t)TOK * DFFN * 2);
  float*    xf   = (float*)   alloc((size_t)TOK * DD * 4);

  // Weight conversion f32 -> f16 with transpose to N-major
  tc_kernel<<<dim3((DD * 3 * DD + 255) / 256, LL), 256, 0, stream>>>(attn_w,  wqT,  DD, 3 * DD);
  tc_kernel<<<dim3((DD * DD + 255) / 256,     LL), 256, 0, stream>>>(attnp_w, wpT,  DD, DD);
  tc_kernel<<<dim3((DD * DFFN + 255) / 256,   LL), 256, 0, stream>>>(fc_w,    wfT,  DD, DFFN);
  tc_kernel<<<dim3((DFFN * DD + 255) / 256,   LL), 256, 0, stream>>>(proj_w,  wprT, DFFN, DD);

  embed_kernel<<<TOK, 256, 0, stream>>>(xs, read_w, read_b, pos_sch, pos_path, hbuf);

  for (int l = 0; l < LL; l++) {
    ln_kernel<<<TOK, 256, 0, stream>>>(hbuf, ln1_g + l * DD, ln1_b + l * DD,
                                       x16, nullptr, 0);
    gemm_wmma<<<dim3((3 * DD) / 64, TOK / 128), 128, 0, stream>>>(
        x16, wqT + (size_t)l * 3 * DD * DD, attn_b + l * 3 * DD,
        qkv16, nullptr, TOK, 3 * DD, DD, 0);
    vt_kernel<<<dim3(SS / 64, BB * HH), 256, 0, stream>>>(qkv16, vT16);
    attn_kernel<<<(BB * HH * (SS / 16)) / 4, 128, 0, stream>>>(
        qkv16, vT16, c16, (l < 2) ? 1 : 0);
    gemm_wmma<<<dim3(DD / 64, TOK / 128), 128, 0, stream>>>(
        c16, wpT + (size_t)l * DD * DD, attnp_b + l * DD,
        nullptr, hbuf, TOK, DD, DD, 2);
    ln_kernel<<<TOK, 256, 0, stream>>>(hbuf, ln2_g + l * DD, ln2_b + l * DD,
                                       x16, nullptr, 0);
    gemm_wmma<<<dim3(DFFN / 64, TOK / 128), 128, 0, stream>>>(
        x16, wfT + (size_t)l * DFFN * DD, fc_b + l * DFFN,
        ff16, nullptr, TOK, DFFN, DD, 1);
    gemm_wmma<<<dim3(DD / 64, TOK / 128), 128, 0, stream>>>(
        ff16, wprT + (size_t)l * DD * DFFN, proj_b + l * DD,
        nullptr, hbuf, TOK, DD, DFFN, 2);
  }

  ln_kernel<<<TOK, 256, 0, stream>>>(hbuf, lnf_g, lnf_b, nullptr, xf, 1);
  head_kernel<<<TOK, 32, 0, stream>>>(xf, out_w, out_b, (float*)d_out);
}
